// Attention_13391708029173
// MI455X (gfx1250) — matrix operand order repaired and checked by reading, not compiled, not run on hardware
//
#include <hip/hip_runtime.h>
#include <hip/hip_bf16.h>

typedef __attribute__((ext_vector_type(16))) __bf16 v16bf;
typedef __attribute__((ext_vector_type(8)))  __bf16 v8bf;
typedef __attribute__((ext_vector_type(4)))  __bf16 v4bf;
typedef __attribute__((ext_vector_type(8)))  float  v8f;
typedef __attribute__((ext_vector_type(4)))  unsigned int v4u;
typedef __attribute__((ext_vector_type(8)))  int v8i_t;
typedef __attribute__((ext_vector_type(4)))  int v4i_t;

#define BB 4
#define NN 2048
#define CC 1024
#define HH 16
#define DD 64
#define MM (BB*NN)          // 8192
#define SCALE 0.125f        // 64^-0.5

// ---------------------------------------------------------------- WMMA helpers

__device__ __forceinline__ v8f wmma_bf16(v16bf a, v16bf b, v8f c) {
  return __builtin_amdgcn_wmma_f32_16x16x32_bf16(false, a, false, b, (short)0, c,
                                                 false, false);
}

// 16-bit A-matrix 16x32 fragment: lane holds 16 contiguous K at
// K = {half*16..+15}. Caller passes p = row + k0 + half*16.
__device__ __forceinline__ v16bf load_a_frag(const __bf16* p) {
  v8bf lo = *(const v8bf*)p;
  v8bf hi = *(const v8bf*)(p + 8);
  return __builtin_shufflevector(lo, hi, 0,1,2,3,4,5,6,7,8,9,10,11,12,13,14,15);
}

// ---------------------------------------------------------------- TDM helpers

// Generic pointer to LDS: the LDS offset is the low 32 bits of the address.
__device__ __forceinline__ unsigned lds_off(const void* p) {
  return (unsigned)(uintptr_t)p;
}

// 2D tile load via Tensor Data Mover (bf16 elements, data_size = 2 bytes).
// Tile of tile1 rows x tile0 contiguous elements from a row-major tensor with
// row stride stride0 (elements); packed row-major into LDS at lds_addr.
__device__ __forceinline__ void tdm_load_2d(unsigned lds_addr, const void* gptr,
                                            unsigned tensor_d0, unsigned tensor_d1,
                                            unsigned stride0,
                                            unsigned tile0, unsigned tile1) {
  unsigned long long ga = (unsigned long long)(uintptr_t)gptr;
  v4u g0;
  g0[0] = 1u;                                             // count=1, user D#
  g0[1] = lds_addr;                                       // lds_addr[31:0]
  g0[2] = (unsigned)ga;                                   // global_addr[31:0]
  g0[3] = (unsigned)((ga >> 32) & 0x01FFFFFFu) | 0x80000000u; // addr[56:32]|type=2
  v8i_t g1;
  g1[0] = (int)(1u << 16);                                // data_size=1 (2B)
  g1[1] = (int)((tensor_d0 & 0xFFFFu) << 16);             // tensor_dim0[15:0]
  g1[2] = (int)((tensor_d0 >> 16) | ((tensor_d1 & 0xFFFFu) << 16));
  g1[3] = (int)((tensor_d1 >> 16) | (tile0 << 16));       // tile_dim0
  g1[4] = (int)(tile1 & 0xFFFFu);                         // tile_dim1, tile_dim2=0
  g1[5] = (int)stride0;                                   // tensor_dim0_stride lo
  g1[6] = 0;                                              // stride hi / dim1_stride
  g1[7] = 0;
  v4i_t gz = {0, 0, 0, 0};                                // 2D: groups 2/3 unused
  v8i_t gz8 = {0, 0, 0, 0, 0, 0, 0, 0};
  __builtin_amdgcn_tensor_load_to_lds(g0, g1, gz, gz, gz8, 0);
}

// ---------------------------------------------------------------- f32 -> bf16

__global__ __launch_bounds__(256) void cvt_f32_to_bf16(const float* __restrict__ src,
                                                       __bf16* __restrict__ dst,
                                                       int n4) {
  int i = blockIdx.x * blockDim.x + threadIdx.x;
  if (i < n4) {
    float4 f = ((const float4*)src)[i];
    v4bf o;
    o[0] = (__bf16)f.x; o[1] = (__bf16)f.y; o[2] = (__bf16)f.z; o[3] = (__bf16)f.w;
    ((v4bf*)dst)[i] = o;
  }
}

// ---------------------------------------------------------------- GEMM core
// Block = 64x64 output, 4 waves; TDM double-buffered A/B tile staging in LDS.
// A = X [*,CC] rows mblk..+63; B column n = W row n (rows nblk..+63).

__device__ __forceinline__ void gemm_core(const __bf16* __restrict__ X,
                                          const __bf16* __restrict__ W,
                                          int mblk, int nblk,
                                          __bf16 (*As)[64 * 128],
                                          __bf16 (*Bs)[64 * 128],
                                          v8f acc[4]) {
  const int wave = threadIdx.x >> 5;
  const int lane = threadIdx.x & 31;
  const int half = lane >> 4;
  const int lrow = lane & 15;
  const bool issuer = (threadIdx.x < 32);

  if (issuer) {
    tdm_load_2d(lds_off(As[0]), X + (size_t)mblk * CC, CC, MM, CC, 128, 64);
    tdm_load_2d(lds_off(Bs[0]), W + (size_t)nblk * CC, CC, CC, CC, 128, 64);
  }

  for (int s = 0; s < 8; ++s) {                    // 8 stages x 128 K
    const int buf = s & 1;
    if (issuer) {
      if (s + 1 < 8) {
        const int k1 = (s + 1) * 128;
        tdm_load_2d(lds_off(As[buf ^ 1]), X + (size_t)mblk * CC + k1,
                    CC, MM, CC, 128, 64);
        tdm_load_2d(lds_off(Bs[buf ^ 1]), W + (size_t)nblk * CC + k1,
                    CC, CC, CC, 128, 64);
        __builtin_amdgcn_s_wait_tensorcnt((short)2);   // stage s landed
      } else {
        __builtin_amdgcn_s_wait_tensorcnt((short)0);
      }
    }
    __syncthreads();

    const __bf16* Arow = As[buf] + (wave * 16 + lrow) * 128;
#pragma unroll
    for (int kk = 0; kk < 128; kk += 32) {
      v16bf a = load_a_frag(Arow + kk + half * 16);
#pragma unroll
      for (int t = 0; t < 4; ++t) {
        const __bf16* Bp = Bs[buf] + (t * 16 + lrow) * 128 + kk + half * 16;
        acc[t] = wmma_bf16(a, *(const v16bf*)Bp, acc[t]);
      }
    }
    __syncthreads();                               // buffer free for next DMA
  }
}

// ---------------------------------------------------------------- QKV GEMM

__global__ __launch_bounds__(128) void gemm_qkv(const __bf16* __restrict__ X,
                                                const __bf16* __restrict__ Wq,
                                                const __bf16* __restrict__ Wk,
                                                const __bf16* __restrict__ Wv,
                                                __bf16* __restrict__ Qo,
                                                __bf16* __restrict__ Ko,
                                                __bf16* __restrict__ Vto) {
  const int blocksPerW = (MM/64) * (CC/64);        // 2048
  const int wsel  = blockIdx.x / blocksPerW;       // 0=q 1=k 2=v
  const int rem   = blockIdx.x % blocksPerW;
  const int mblk  = (rem / (CC/64)) * 64;
  const int nblk  = (rem % (CC/64)) * 64;
  const __bf16* W = (wsel == 0) ? Wq : ((wsel == 1) ? Wk : Wv);

  __shared__ alignas(128) __bf16 As[2][64 * 128];  // 32 KB
  __shared__ alignas(128) __bf16 Bs[2][64 * 128];  // 32 KB

  v8f acc[4] = {};
  gemm_core(X, W, mblk, nblk, As, Bs, acc);

  const int wave = threadIdx.x >> 5;
  const int lane = threadIdx.x & 31;
  const int half = lane >> 4;
  const int lrow = lane & 15;
  const int m0 = mblk + wave * 16;

#pragma unroll
  for (int t = 0; t < 4; ++t) {
#pragma unroll
    for (int j = 0; j < 8; ++j) {
      int mIdx = m0 + j + 8 * half;
      int cIdx = nblk + t * 16 + lrow;
      int bb = mIdx >> 11, nseq = mIdx & (NN - 1);
      int h  = cIdx >> 6,  d    = cIdx & (DD - 1);
      float f = acc[t][j];
      if (wsel == 0)
        Qo[(((size_t)bb * HH + h) * NN + nseq) * DD + d] = (__bf16)(f * SCALE);
      else if (wsel == 1)
        Ko[(((size_t)bb * HH + h) * NN + nseq) * DD + d] = (__bf16)f;
      else
        Vto[(((size_t)bb * HH + h) * DD + d) * NN + nseq] = (__bf16)f;  // V^T
    }
  }
}

// ---------------------------------------------------------------- flash attention
// grid.x = B*H * (N/64); 128 threads = 4 waves, each wave owns 16 query rows.

__global__ __launch_bounds__(128) void flash_attn(const __bf16* __restrict__ Q,
                                                  const __bf16* __restrict__ K,
                                                  const __bf16* __restrict__ Vt,
                                                  __bf16* __restrict__ Out) {
  const int bh    = blockIdx.x >> 5;               // 0..63
  const int qtile = blockIdx.x & 31;
  const __bf16* Qb = Q  + (size_t)bh * NN * DD;
  const __bf16* Kb = K  + (size_t)bh * NN * DD;
  const __bf16* Vb = Vt + (size_t)bh * DD * NN;

  const int wave = threadIdx.x >> 5;
  const int lane = threadIdx.x & 31;
  const int half = lane >> 4;
  const int lrow = lane & 15;
  const int q0   = qtile * 64 + wave * 16;

  __shared__ alignas(128) __bf16 Ks[64 * 64];      // [key][d]
  __shared__ alignas(128) __bf16 Vs[64 * 64];      // [d][key]   (V^T tile)
  __shared__ alignas(128) __bf16 Ps[4][16 * 64];   // per-wave P [m][key]

  const __bf16* Qrow = Qb + (size_t)(q0 + lrow) * DD;
  v16bf qa0 = load_a_frag(Qrow + 0  + half * 16);
  v16bf qa1 = load_a_frag(Qrow + 32 + half * 16);

  v8f o[4] = {};
  float rmax[8], rsum[8];
#pragma unroll
  for (int j = 0; j < 8; ++j) { rmax[j] = -3.0e38f; rsum[j] = 0.0f; }

  for (int kb = 0; kb < NN; kb += 64) {
    __syncthreads();                               // prior reads of Ks/Vs done
    if (threadIdx.x < 32) {
      // K tile: 64 keys x 64 d from [NN, DD] row-major at row kb
      tdm_load_2d(lds_off(Ks), Kb + (size_t)kb * DD, DD, NN, DD, 64, 64);
      // V^T tile: 64 d-rows x 64 keys from [DD, NN] row-major at col kb
      tdm_load_2d(lds_off(Vs), Vb + kb, NN, DD, NN, 64, 64);
      __builtin_amdgcn_s_wait_tensorcnt((short)0);
    }
    __syncthreads();

    // S = Q K^T : 4 key-tiles x (2 WMMAs over d)
    v8f s[4];
#pragma unroll
    for (int t = 0; t < 4; ++t) {
      v8f a = {};
      const __bf16* base = Ks + (t * 16 + lrow) * 64;
      a = wmma_bf16(qa0, *(const v16bf*)(base + 0  + half * 16), a);
      a = wmma_bf16(qa1, *(const v16bf*)(base + 32 + half * 16), a);
      s[t] = a;
    }

    // online softmax (rows wave-local; reduce across the 16 lanes of a half)
#pragma unroll
    for (int j = 0; j < 8; ++j) {
      float mx = fmaxf(fmaxf(s[0][j], s[1][j]), fmaxf(s[2][j], s[3][j]));
      mx = fmaxf(mx, __shfl_xor(mx, 1, 32));
      mx = fmaxf(mx, __shfl_xor(mx, 2, 32));
      mx = fmaxf(mx, __shfl_xor(mx, 4, 32));
      mx = fmaxf(mx, __shfl_xor(mx, 8, 32));
      float newm = fmaxf(rmax[j], mx);
      float corr = __expf(rmax[j] - newm);
      rmax[j] = newm;
      float ssum = 0.0f;
#pragma unroll
      for (int t = 0; t < 4; ++t) {
        float p = __expf(s[t][j] - newm);
        s[t][j] = p;
        ssum += p;
      }
      ssum += __shfl_xor(ssum, 1, 32);
      ssum += __shfl_xor(ssum, 2, 32);
      ssum += __shfl_xor(ssum, 4, 32);
      ssum += __shfl_xor(ssum, 8, 32);
      rsum[j] = rsum[j] * corr + ssum;
#pragma unroll
      for (int t = 0; t < 4; ++t) o[t][j] *= corr;
    }

    // P -> LDS (bf16, [m][key]) to rebuild A-fragment layout
#pragma unroll
    for (int t = 0; t < 4; ++t)
#pragma unroll
      for (int j = 0; j < 8; ++j)
        Ps[wave][(j + 8 * half) * 64 + t * 16 + lrow] = (__bf16)s[t][j];
    asm volatile("s_wait_dscnt 0x0" ::: "memory");   // wave-local LDS RAW

    const __bf16* Prow = &Ps[wave][lrow * 64];
    v16bf pa0 = load_a_frag(Prow + 0  + half * 16);
    v16bf pa1 = load_a_frag(Prow + 32 + half * 16);

    // O += P V : 4 d-tiles x (2 WMMAs over keys)
#pragma unroll
    for (int t = 0; t < 4; ++t) {
      const __bf16* base = Vs + (t * 16 + lrow) * 64;
      o[t] = wmma_bf16(pa0, *(const v16bf*)(base + 0  + half * 16), o[t]);
      o[t] = wmma_bf16(pa1, *(const v16bf*)(base + 32 + half * 16), o[t]);
    }
  }

  // epilogue: normalize, write attn-out as bf16 [B,N,C] with c = h*64+d
  const int b = bh >> 4, h = bh & 15;
#pragma unroll
  for (int t = 0; t < 4; ++t)
#pragma unroll
    for (int j = 0; j < 8; ++j) {
      int nseq = q0 + j + 8 * half;
      int d    = t * 16 + lrow;
      float v  = o[t][j] / rsum[j];
      Out[((size_t)b * NN + nseq) * CC + h * DD + d] = (__bf16)v;
    }
}

// ---------------------------------------------------------------- output GEMM

__global__ __launch_bounds__(128) void gemm_out(const __bf16* __restrict__ A,
                                                const __bf16* __restrict__ Wo,
                                                const float* __restrict__ bias,
                                                float* __restrict__ out) {
  const int mblk = (blockIdx.x / (CC/64)) * 64;
  const int nblk = (blockIdx.x % (CC/64)) * 64;

  __shared__ alignas(128) __bf16 As[2][64 * 128];
  __shared__ alignas(128) __bf16 Bs[2][64 * 128];

  v8f acc[4] = {};
  gemm_core(A, Wo, mblk, nblk, As, Bs, acc);

  const int wave = threadIdx.x >> 5;
  const int lane = threadIdx.x & 31;
  const int half = lane >> 4;
  const int lrow = lane & 15;
  const int m0 = mblk + wave * 16;

#pragma unroll
  for (int t = 0; t < 4; ++t)
#pragma unroll
    for (int j = 0; j < 8; ++j) {
      int mIdx = m0 + j + 8 * half;
      int cIdx = nblk + t * 16 + lrow;
      out[(size_t)mIdx * CC + cIdx] = acc[t][j] + bias[cIdx];
    }
}

// ---------------------------------------------------------------- launch

extern "C" void kernel_launch(void* const* d_in, const int* in_sizes, int n_in,
                              void* d_out, int out_size, void* d_ws, size_t ws_size,
                              hipStream_t stream) {
  const float* x  = (const float*)d_in[0];
  const float* wq = (const float*)d_in[1];
  const float* wk = (const float*)d_in[2];
  const float* wv = (const float*)d_in[3];
  const float* wo = (const float*)d_in[4];
  const float* bo = (const float*)d_in[5];

  char* ws = (char*)d_ws;
  // X16 (16MB, reused as attn-out) | W16 x4 (8MB) | Q (16MB) | K (16MB) | Vt (16MB)
  __bf16* X16  = (__bf16*)ws;
  __bf16* Wq16 = (__bf16*)(ws + ((size_t)16 << 20));
  __bf16* Wk16 = Wq16 + (size_t)CC * CC;
  __bf16* Wv16 = Wk16 + (size_t)CC * CC;
  __bf16* Wo16 = Wv16 + (size_t)CC * CC;
  __bf16* Qb   = (__bf16*)(ws + ((size_t)24 << 20));
  __bf16* Kb   = (__bf16*)(ws + ((size_t)40 << 20));
  __bf16* Vt   = (__bf16*)(ws + ((size_t)56 << 20));
  __bf16* ATT16 = X16;   // safe: consumed by gemm_qkv before flash_attn writes it

  const int nx4 = MM * CC / 4;
  const int nw4 = CC * CC / 4;
  cvt_f32_to_bf16<<<nx4 / 256, 256, 0, stream>>>(x,  X16,  nx4);
  cvt_f32_to_bf16<<<nw4 / 256, 256, 0, stream>>>(wq, Wq16, nw4);
  cvt_f32_to_bf16<<<nw4 / 256, 256, 0, stream>>>(wk, Wk16, nw4);
  cvt_f32_to_bf16<<<nw4 / 256, 256, 0, stream>>>(wv, Wv16, nw4);
  cvt_f32_to_bf16<<<nw4 / 256, 256, 0, stream>>>(wo, Wo16, nw4);

  gemm_qkv<<<3 * (MM/64) * (CC/64), 128, 0, stream>>>(X16, Wq16, Wk16, Wv16,
                                                      Qb, Kb, Vt);
  flash_attn<<<BB * HH * (NN/64), 128, 0, stream>>>(Qb, Kb, Vt, ATT16);
  gemm_out<<<(MM/64) * (CC/64), 128, 0, stream>>>(ATT16, Wo16, bo, (float*)d_out);
}
